// Block_45724221833665
// MI455X (gfx1250) — matrix operand order repaired and checked by reading, not compiled, not run on hardware
//
#include <hip/hip_runtime.h>
#include <hip/hip_bf16.h>
#include <math.h>

// ---------------------------------------------------------------------------
// MI455X / gfx1250 implementation of the gated prefix-attention transformer
// block.  All GEMM-shaped work goes through v_wmma_f32_16x16x32_f16 (wave32,
// f16 inputs, f32 accumulate).  GEMM operands are double-buffered in LDS via
// CDNA5 async global->LDS copies (GLOBAL_LOAD_ASYNC_TO_LDS_B128) with a
// non-zero s_wait_asynccnt threshold so the DMA of tile k+1 overlaps the
// WMMA work on tile k.  The final K-step is peeled so the steady-state loop
// body is branch-free.  Key simplification: softmax over a length-1 slice
// == 1, so attention column 0 is the constant tanh(gate[h]).
// ---------------------------------------------------------------------------

typedef __attribute__((ext_vector_type(16))) _Float16 v16h;
typedef __attribute__((ext_vector_type(8)))  _Float16 f16x8;
typedef __attribute__((ext_vector_type(8)))  float    v8f;

union V16 { v16h v; f16x8 h[2]; };

__device__ inline v8f wmma_f16(v16h a, v16h b, v8f c) {
  // (neg_a, A, neg_b, B, c_mod, C, reuse_a, reuse_b)
  return __builtin_amdgcn_wmma_f32_16x16x32_f16(false, a, false, b, (short)0, c,
                                                false, false);
}

// Async 16B copy global -> LDS.  The VDST operand is a VGPR holding the
// wave-relative LDS byte address (== low 32 bits of the generic pointer,
// per the CDNA5 flat-aperture mapping addr[31:0] -> LDS offset).
__device__ inline void async_load_b128(unsigned int lds_addr, const void* g) {
  asm volatile("global_load_async_to_lds_b128 %0, %1, off"
               :: "v"(lds_addr), "v"(g) : "memory");
}
// Wait until at most N async copies remain in flight (in-order completion).
#if defined(__has_builtin) && __has_builtin(__builtin_amdgcn_s_wait_asynccnt)
#define WAIT_ASYNC(n) __builtin_amdgcn_s_wait_asynccnt(n)
#else
#define WAIT_ASYNC(n) asm volatile("s_wait_asynccnt %0" :: "i"(n) : "memory")
#endif

#define BB   8
#define NN   1024
#define PP   64
#define DD   1024
#define HH   16
#define HDIM 64
#define HID_ 4096
#define MTOT 1088   // P + N

// ---------------------------------------------------------------------------
// Weight transpose + f32->f16:  in (K_ x N_) row-major  ->  out (N_ x K_)
// ---------------------------------------------------------------------------
__global__ void k_transpose_w(const float* __restrict__ in,
                              _Float16* __restrict__ out, int K_, int N_) {
  long idx = (long)blockIdx.x * blockDim.x + threadIdx.x;
  long total = (long)K_ * N_;
  if (idx >= total) return;
  int n = (int)(idx / K_);
  int k = (int)(idx % K_);
  out[idx] = (_Float16)in[(long)k * N_ + n];
}

// ---------------------------------------------------------------------------
// LayerNorm: one 256-thread block per row.  in is f32 or f16, out f16.
// Supports in-place (f16 -> f16) since writes follow the final read.
// ---------------------------------------------------------------------------
__global__ void k_layernorm(const void* in, int inF16, _Float16* out,
                            const float* __restrict__ g,
                            const float* __restrict__ b, int D_) {
  const int row = blockIdx.x;
  const int tid = threadIdx.x;
  const float*    inF = (const float*)in;
  const _Float16* inH = (const _Float16*)in;
  const long base = (long)row * D_;
  float s = 0.f, s2 = 0.f;
  for (int c = tid; c < D_; c += 256) {
    float v = inF16 ? (float)inH[base + c] : inF[base + c];
    s += v; s2 += v * v;
  }
  __shared__ float rs[256], rq[256];
  rs[tid] = s; rq[tid] = s2;
  __syncthreads();
  for (int off = 128; off > 0; off >>= 1) {
    if (tid < off) { rs[tid] += rs[tid + off]; rq[tid] += rq[tid + off]; }
    __syncthreads();
  }
  const float mean = rs[0] / (float)D_;
  const float var  = rq[0] / (float)D_ - mean * mean;
  const float inv  = rsqrtf(var + 1e-5f);
  for (int c = tid; c < D_; c += 256) {
    float v = inF16 ? (float)inH[base + c] : inF[base + c];
    out[base + c] = (_Float16)((v - mean) * inv * g[c] + b[c]);
  }
}

// ---------------------------------------------------------------------------
// Generic WMMA GEMM:  C(M x N) = A(M x K, f16 row-major) * BT(N x K, f16)^T
// 64x128 macro-tile per 128-thread block; 4 waves, each owning 2x4 16x16
// tiles (8 v_wmma per K-step).  BK = 32.  Double-buffered async global->LDS
// staging: 6 async b128 copies per thread per tile; while tile k is consumed
// by the matrix pipe, tile k+1 streams into the other LDS buffer
// (s_wait_asynccnt 6 = "oldest tile landed, newest still in flight").
// The last K-step is peeled so the hot loop is branch-free.
// flags: 1=bias  2=gelu  4=residual(res, stride N)  8=f32 out  16=V^T store
// dest row mapping: grow -> b = grow/rowsPerB, r = grow%rowsPerB
//   natural: out[(b*dRowsPerB + dOff + r)*ldd + col]
//   vT     : out[((b*H + col/64)*64 + col%64)*MtotVT + dOff + r]
// ---------------------------------------------------------------------------
__global__ __launch_bounds__(128) void k_gemm_wmma(
    const _Float16* __restrict__ A, const _Float16* __restrict__ BT,
    const float* __restrict__ bias, const float* __restrict__ res,
    void* __restrict__ out, int M, int N, int K,
    int rowsPerB, int dRowsPerB, int dOff, int ldd, int flags, int MtotVT) {
  __shared__ __align__(16) _Float16 As[2][64 * 32];    //  8 KB
  __shared__ __align__(16) _Float16 Bs[2][128 * 32];   // 16 KB

  const int mbase = blockIdx.x * 64;
  const int nbase = blockIdx.y * 128;
  const int tid  = threadIdx.x;
  const int wave = tid >> 5;
  const int lane = tid & 31;
  const int lr = lane & 15;          // row-in-tile / col-in-tile
  const int lh = lane >> 4;          // K-half selector
  const int wr = wave >> 1;          // wave row (0..1): 32 rows each
  const int wc = wave & 1;           // wave col (0..1): 64 cols each

  // wave-relative LDS byte addresses of the two staging buffers
  const unsigned int asB = (unsigned int)(unsigned long long)&As[0][0];
  const unsigned int bsB = (unsigned int)(unsigned long long)&Bs[0][0];

  // A tile: 64x32 f16 = 256 16B chunks (2/thread); B tile: 128x32 = 512 (4/thread)
  const int pa = tid & 3;                  // 16B chunk within a 32-f16 row
  const int ra0 = tid >> 2, ra1 = (tid >> 2) + 32;
  const unsigned int dA0 = (unsigned int)(ra0 * 32 + pa * 8) * 2u;
  const unsigned int dA1 = (unsigned int)(ra1 * 32 + pa * 8) * 2u;
  const _Float16* gA0 = A + (size_t)(mbase + ra0) * K + pa * 8;
  const _Float16* gA1 = A + (size_t)(mbase + ra1) * K + pa * 8;
  unsigned int dB[4];
  const _Float16* gB[4];
#pragma unroll
  for (int j = 0; j < 4; ++j) {
    const int rb = (tid >> 2) + 32 * j;
    dB[j] = (unsigned int)(rb * 32 + pa * 8) * 2u;
    gB[j] = BT + (size_t)(nbase + rb) * K + pa * 8;
  }

  v8f acc[2][4] = {};
  const int nk = K >> 5;

  // issue one tile's 6 async copies into buffer `buf` and advance pointers
  auto issue_tile = [&](int buf) {
    const unsigned int ao = asB + (buf ? 64 * 32 * 2 : 0);
    const unsigned int bo = bsB + (buf ? 128 * 32 * 2 : 0);
    async_load_b128(ao + dA0, gA0);
    async_load_b128(ao + dA1, gA1);
#pragma unroll
    for (int j = 0; j < 4; ++j) async_load_b128(bo + dB[j], gB[j]);
    gA0 += 32; gA1 += 32;
#pragma unroll
    for (int j = 0; j < 4; ++j) gB[j] += 32;
  };

  // consume one staged tile: 12 ds_load_b128 fragment reads + 8 v_wmma
  auto compute_tile = [&](int buf) {
    const _Float16* as = &As[buf][0];
    const _Float16* bs = &Bs[buf][0];
    V16 aF[2], bF[4];
#pragma unroll
    for (int ii = 0; ii < 2; ++ii) {
      const int row = wr * 32 + ii * 16 + lr;
      aF[ii].h[0] = *(const f16x8*)&as[row * 32 + lh * 8];
      aF[ii].h[1] = *(const f16x8*)&as[row * 32 + 16 + lh * 8];
    }
#pragma unroll
    for (int jj = 0; jj < 4; ++jj) {
      const int row = wc * 64 + jj * 16 + lr;
      bF[jj].h[0] = *(const f16x8*)&bs[row * 32 + lh * 8];
      bF[jj].h[1] = *(const f16x8*)&bs[row * 32 + 16 + lh * 8];
    }
#pragma unroll
    for (int ii = 0; ii < 2; ++ii)
#pragma unroll
      for (int jj = 0; jj < 4; ++jj)
        acc[ii][jj] = wmma_f16(aF[ii].v, bF[jj].v, acc[ii][jj]);
  };

  // ---- prologue: stream tile 0 into buffer 0
  issue_tile(0);

  // ---- steady state (branch-free): overlap DMA of tile kk+1 with tile kk
  for (int kk = 0; kk < nk - 1; ++kk) {
    issue_tile((kk + 1) & 1);
    __builtin_prefetch(gA0, 0, 0);     // speculative touch of tile kk+2
    __builtin_prefetch(gB[0], 0, 0);
    WAIT_ASYNC(6);                     // tile kk landed; kk+1 still in flight
    __syncthreads();                   // publish tile kk to all 4 waves
    compute_tile(kk & 1);
    __syncthreads();                   // protect buffer tile kk+2 overwrites
  }

  // ---- peeled final K-step
  WAIT_ASYNC(0);
  __syncthreads();
  compute_tile((nk - 1) & 1);

  // ---- epilogue
#pragma unroll
  for (int ii = 0; ii < 2; ++ii) {
#pragma unroll
    for (int jj = 0; jj < 4; ++jj) {
      const int gcol = nbase + wc * 64 + jj * 16 + lr;
#pragma unroll
      for (int r = 0; r < 8; ++r) {
        const int grow = mbase + wr * 32 + ii * 16 + r + 8 * lh;
        float v = acc[ii][jj][r];
        if (flags & 1) v += bias[gcol];
        if (flags & 2) v = 0.5f * v * (1.0f + erff(v * 0.70710678118f));
        if (flags & 4) v += res[(size_t)grow * N + gcol];
        const int bidx = grow / rowsPerB;
        const int rib  = grow % rowsPerB;
        size_t idx;
        if (flags & 16) {   // V^T layout: [b][h][d][m]
          idx = (((size_t)bidx * HH + gcol / HDIM) * HDIM + (gcol % HDIM)) *
                    (size_t)MtotVT + dOff + rib;
        } else {
          idx = ((size_t)bidx * dRowsPerB + dOff + rib) * (size_t)ldd + gcol;
        }
        if (flags & 8) ((float*)out)[idx] = v;
        else           ((_Float16*)out)[idx] = (_Float16)v;
      }
    }
  }
}

// ---------------------------------------------------------------------------
// Fused attention: one block per (b*H, 32-query tile); 2 waves, 16 rows each.
// scores (f32) and probs (f16) live in dynamic LDS (208896 B < 320 KB/WGP).
// Column 0 weight is the constant tanh(gate[h]); softmax spans keys 1..1087.
// ---------------------------------------------------------------------------
__global__ __launch_bounds__(64) void k_attention(
    const _Float16* __restrict__ qh, const _Float16* __restrict__ kc,
    const _Float16* __restrict__ vt, const float* __restrict__ gate,
    float* __restrict__ o) {
  extern __shared__ char smem[];
  const int bh = blockIdx.x;
  const int b = bh >> 4, h = bh & 15;
  const int tid = threadIdx.x;
  const int w = tid >> 5, lane = tid & 31;
  const int lr = lane & 15, lh = lane >> 4;
  const int n0 = blockIdx.y * 32 + w * 16;

  float*    sS = (float*)smem + (size_t)w * 16 * MTOT;
  _Float16* sP = (_Float16*)(smem + 2 * 16 * MTOT * 4) + (size_t)w * 16 * MTOT;

  // Q A-fragments (reused over all 68 key tiles), K-dim = head dim (64 = 2x32)
  const _Float16* qb = qh + ((size_t)b * NN + n0 + lr) * DD + h * HDIM;
  V16 aQ[2];
#pragma unroll
  for (int s = 0; s < 2; ++s) {
    aQ[s].h[0] = *(const f16x8*)(qb + s * 32 + lh * 8);
    aQ[s].h[1] = *(const f16x8*)(qb + s * 32 + 16 + lh * 8);
  }

  // ---- scores = Q K^T / 8 into LDS
  for (int mt = 0; mt < MTOT / 16; ++mt) {
    const _Float16* kb = kc + ((size_t)b * MTOT + mt * 16 + lr) * DD + h * HDIM;
    V16 bK0, bK1;
    bK0.h[0] = *(const f16x8*)(kb + lh * 8);
    bK0.h[1] = *(const f16x8*)(kb + 16 + lh * 8);
    bK1.h[0] = *(const f16x8*)(kb + 32 + lh * 8);
    bK1.h[1] = *(const f16x8*)(kb + 32 + 16 + lh * 8);
    v8f acc = {};
    acc = wmma_f16(aQ[0].v, bK0.v, acc);
    acc = wmma_f16(aQ[1].v, bK1.v, acc);
#pragma unroll
    for (int r = 0; r < 8; ++r)
      sS[(r + 8 * lh) * MTOT + mt * 16 + lr] = acc[r] * 0.125f;
  }

  // ---- per-row softmax over keys 1..MTOT-1; column 0 := tanh(gate[h])
  const float wp = tanhf(gate[h]);
  for (int r = 0; r < 16; ++r) {
    float* row = sS + r * MTOT;
    float mx = -3.0e38f;
    for (int c = 1 + lane; c < MTOT; c += 32) mx = fmaxf(mx, row[c]);
#pragma unroll
    for (int m = 16; m; m >>= 1) mx = fmaxf(mx, __shfl_xor(mx, m, 32));
    float sm = 0.f;
    for (int c = 1 + lane; c < MTOT; c += 32) sm += __expf(row[c] - mx);
#pragma unroll
    for (int m = 16; m; m >>= 1) sm += __shfl_xor(sm, m, 32);
    const float rinv = 1.0f / sm;
    _Float16* prow = sP + r * MTOT;
    if (lane == 0) prow[0] = (_Float16)wp;
    for (int c = 1 + lane; c < MTOT; c += 32)
      prow[c] = (_Float16)(__expf(row[c] - mx) * rinv);
  }

  // ---- O = P V  (A = probs from LDS, B = V^T [b][h][d][m] from global)
  v8f accO[4] = {};
  for (int km = 0; km < MTOT / 32; ++km) {
    V16 aP;
    const _Float16* pb = sP + (size_t)lr * MTOT + km * 32;
    aP.h[0] = *(const f16x8*)(pb + lh * 8);
    aP.h[1] = *(const f16x8*)(pb + 16 + lh * 8);
#pragma unroll
    for (int t = 0; t < 4; ++t) {
      const _Float16* vb =
          vt + ((size_t)(b * HH + h) * HDIM + t * 16 + lr) * MTOT + km * 32 + lh * 8;
      V16 bV;
      bV.h[0] = *(const f16x8*)vb;
      bV.h[1] = *(const f16x8*)(vb + 16);
      accO[t] = wmma_f16(aP.v, bV.v, accO[t]);
    }
  }
#pragma unroll
  for (int t = 0; t < 4; ++t)
#pragma unroll
    for (int r = 0; r < 8; ++r)
      o[((size_t)b * NN + n0 + r + 8 * lh) * DD + h * HDIM + t * 16 + lr] =
          accO[t][r];
}

// ---------------------------------------------------------------------------
// Host-side orchestration
// ---------------------------------------------------------------------------
extern "C" void kernel_launch(void* const* d_in, const int* in_sizes, int n_in,
                              void* d_out, int out_size, void* d_ws,
                              size_t ws_size, hipStream_t stream) {
  const float* x      = (const float*)d_in[0];
  const float* x_text = (const float*)d_in[1];
  const float* n1_g   = (const float*)d_in[2];
  const float* n1_b   = (const float*)d_in[3];
  const float* Wq     = (const float*)d_in[4];
  const float* bq     = (const float*)d_in[5];
  const float* Wk     = (const float*)d_in[6];
  const float* Wv     = (const float*)d_in[7];
  const float* bv     = (const float*)d_in[8];
  const float* ln_g   = (const float*)d_in[9];
  const float* ln_b   = (const float*)d_in[10];
  const float* Wp     = (const float*)d_in[11];
  const float* bp     = (const float*)d_in[12];
  const float* gate   = (const float*)d_in[13];
  const float* n2_g   = (const float*)d_in[14];
  const float* n2_b   = (const float*)d_in[15];
  const float* n3_g   = (const float*)d_in[16];
  const float* n3_b   = (const float*)d_in[17];
  const float* W1     = (const float*)d_in[18];
  const float* b1     = (const float*)d_in[19];
  const float* ffn_g  = (const float*)d_in[20];
  const float* ffn_b  = (const float*)d_in[21];
  const float* W2     = (const float*)d_in[22];
  const float* b2     = (const float*)d_in[23];
  float* out = (float*)d_out;

  const int RN = BB * NN;   // 8192 rows of x
  const int RP = BB * PP;   // 512 rows of x_text

  // ---- workspace carve-up (all f16 regions are 16B aligned by construction)
  _Float16* p = (_Float16*)d_ws;
  _Float16* wqT = p; p += (size_t)DD * DD;
  _Float16* wkT = p; p += (size_t)DD * DD;
  _Float16* wvT = p; p += (size_t)DD * DD;
  _Float16* wpT = p; p += (size_t)DD * DD;
  _Float16* w1T = p; p += (size_t)DD * HID_;
  _Float16* w2T = p; p += (size_t)DD * HID_;
  _Float16* x1h = p; p += (size_t)RN * DD;
  _Float16* xth = p; p += (size_t)RP * DD;
  _Float16* qh  = p; p += (size_t)RN * DD;
  _Float16* kc  = p; p += (size_t)BB * MTOT * DD;
  _Float16* vt  = p; p += (size_t)BB * MTOT * DD;
  _Float16* hbuf= p; p += (size_t)RN * HID_;
  float* obuf = (float*)p;
  float* xres = obuf + (size_t)RN * DD;
  _Float16* oln = x1h;   // reuse: x1h dead after k/v GEMMs
  _Float16* x2h = qh;    // reuse: qh dead after attention

  // ---- 1. weights -> f16 transposed
  {
    int t1 = DD * DD, t2 = DD * HID_;
    k_transpose_w<<<(t1 + 255) / 256, 256, 0, stream>>>(Wq, wqT, DD, DD);
    k_transpose_w<<<(t1 + 255) / 256, 256, 0, stream>>>(Wk, wkT, DD, DD);
    k_transpose_w<<<(t1 + 255) / 256, 256, 0, stream>>>(Wv, wvT, DD, DD);
    k_transpose_w<<<(t1 + 255) / 256, 256, 0, stream>>>(Wp, wpT, DD, DD);
    k_transpose_w<<<(t2 + 255) / 256, 256, 0, stream>>>(W1, w1T, DD, HID_);
    k_transpose_w<<<(t2 + 255) / 256, 256, 0, stream>>>(W2, w2T, HID_, DD);
  }

  // ---- 2. pre-norms
  k_layernorm<<<RN, 256, 0, stream>>>(x, 0, x1h, n1_g, n1_b, DD);
  k_layernorm<<<RP, 256, 0, stream>>>(x_text, 0, xth, n3_g, n3_b, DD);

  // ---- 3. Q, K, V projections  (grid = M/64 x N/128)
  // q = x1 @ Wq + bq  -> qh [b][n][d]
  k_gemm_wmma<<<dim3(RN / 64, DD / 128), 128, 0, stream>>>(
      x1h, wqT, bq, nullptr, qh, RN, DD, DD, NN, NN, 0, DD, /*flags*/1, 0);
  // k_text -> kc rows [0,64)
  k_gemm_wmma<<<dim3(RP / 64, DD / 128), 128, 0, stream>>>(
      xth, wkT, nullptr, nullptr, kc, RP, DD, DD, PP, MTOT, 0, DD, 0, 0);
  // k_main -> kc rows [64,1088)
  k_gemm_wmma<<<dim3(RN / 64, DD / 128), 128, 0, stream>>>(
      x1h, wkT, nullptr, nullptr, kc, RN, DD, DD, NN, MTOT, PP, DD, 0, 0);
  // v_text -> vt [b][h][d][m], m in [0,64)
  k_gemm_wmma<<<dim3(RP / 64, DD / 128), 128, 0, stream>>>(
      xth, wvT, bv, nullptr, vt, RP, DD, DD, PP, 0, 0, 0, 1 | 16, MTOT);
  // v_main -> vt, m in [64,1088)
  k_gemm_wmma<<<dim3(RN / 64, DD / 128), 128, 0, stream>>>(
      x1h, wvT, bv, nullptr, vt, RN, DD, DD, NN, 0, PP, 0, 1 | 16, MTOT);

  // ---- 4. fused gated-softmax attention -> obuf (f32, [b][n][D])
  {
    const size_t smem = (size_t)2 * 16 * MTOT * 4 + (size_t)2 * 16 * MTOT * 2;
    k_attention<<<dim3(BB * HH, NN / 32), 64, smem, stream>>>(qh, kc, vt, gate,
                                                              obuf);
  }

  // ---- 5. inner attn LN, output proj (+bias, +residual x) -> xres (f32)
  k_layernorm<<<RN, 256, 0, stream>>>(obuf, 0, oln, ln_g, ln_b, DD);
  k_gemm_wmma<<<dim3(RN / 64, DD / 128), 128, 0, stream>>>(
      oln, wpT, bp, x, xres, RN, DD, DD, NN, NN, 0, DD, 1 | 4 | 8, 0);

  // ---- 6. MLP branch
  k_layernorm<<<RN, 256, 0, stream>>>(xres, 0, x2h, n2_g, n2_b, DD);
  // h = gelu(x2 @ W1 + b1)  (f16)
  k_gemm_wmma<<<dim3(RN / 64, HID_ / 128), 128, 0, stream>>>(
      x2h, w1T, b1, nullptr, hbuf, RN, HID_, DD, NN, NN, 0, HID_, 1 | 2, 0);
  // ffn_ln in place over 4096
  k_layernorm<<<RN, 256, 0, stream>>>(hbuf, 1, hbuf, ffn_g, ffn_b, HID_);
  // out = xres + h @ W2 + b2  (f32 -> d_out)
  k_gemm_wmma<<<dim3(RN / 64, DD / 128), 128, 0, stream>>>(
      hbuf, w2T, b2, xres, out, RN, DD, HID_, NN, NN, 0, DD, 1 | 4 | 8, 0);
}
